// MultiHeadSelfAttention_1520418422904
// MI455X (gfx1250) — matrix operand order repaired and checked by reading, not compiled, not run on hardware
//
#include <hip/hip_runtime.h>

#define HID 1024
#define NH 16
#define DKD 64
#define BB 4
#define SS 2048
#define ROWS (BB * SS)   // 8192

typedef __attribute__((ext_vector_type(16))) _Float16 v16h;
typedef __attribute__((ext_vector_type(8)))  _Float16 v8h;
typedef __attribute__((ext_vector_type(8)))  float    v8f;

__device__ __forceinline__ v16h frag_combine(v8h lo, v8h hi) {
    v16h r;
#pragma unroll
    for (int i = 0; i < 8; ++i) { r[i] = lo[i]; r[8 + i] = hi[i]; }
    return r;
}

__device__ __forceinline__ v16h load_a_frag(const _Float16* row, int k, int half) {
    // A fragment in contiguous k order (same as B): per lane, K = half*16 + [0..15]
    return frag_combine(*(const v8h*)(row + k + half * 16),
                        *(const v8h*)(row + k + half * 16 + 8));
}

__device__ __forceinline__ v16h load_b_frag(const _Float16* colbase) {
    // B fragment in contiguous k order (same as A): per lane, K = half*16 + [0..15]
    return frag_combine(*(const v8h*)colbase, *(const v8h*)(colbase + 8));
}

__device__ __forceinline__ v8f wmma_f16(v16h a, v16h b, v8f c) {
    return __builtin_amdgcn_wmma_f32_16x16x32_f16(
        false, a, false, b, (short)0, c, false, false);
}

// ---------------------------------------------------------------- convert
__global__ void cvt_f32_to_f16(const float* __restrict__ in,
                               _Float16* __restrict__ out, int n) {
    for (int i = blockIdx.x * blockDim.x + threadIdx.x; i < n;
         i += blockDim.x * gridDim.x)
        out[i] = (_Float16)in[i];
}

// ---------------------------------------------------------------- GEMM
// y[row, n] = sum_k A[row,k] * W[n,k] + bias[n]
// Wave computes a 32(M) x 64(N) tile: 8 accumulators, 2 A-frags + 4 B-frags
// per K=32 step -> 8 WMMAs per 6 fragment loads.
// mode 0: f16 out [b,h,s,dk]   (Q, K)
// mode 1: f16 out [b,h,dk,s]   (V transposed)
// mode 2: f32 out [row, n]     (final projection)
__global__ __launch_bounds__(256) void gemm_wmma(
    const _Float16* __restrict__ A,    // [ROWS, HID]
    const _Float16* __restrict__ W,    // [HID, HID] row-major [e][d]
    const float* __restrict__ bias,    // [HID]
    _Float16* __restrict__ out_h,
    float* __restrict__ out_f,
    int mode) {
    const int lane = threadIdx.x & 31;
    const int wid  = threadIdx.x >> 5;
    const int wg   = blockIdx.x * 8 + wid;     // 4096 waves total
    const int mt   = wg >> 4;                  // 256 m-tiles of 32 rows
    const int ns   = wg & 15;                  // 16 n-strips of 64 cols
    const int half = lane >> 4;
    const int nl   = lane & 15;

    const _Float16* arow0 = A + (size_t)(mt * 32 + nl) * HID;
    const _Float16* arow1 = arow0 + (size_t)16 * HID;

    v8f acc[2][4] = {};
    for (int k = 0; k < HID; k += 32) {
        v16h a0 = load_a_frag(arow0, k, half);
        v16h a1 = load_a_frag(arow1, k, half);
#pragma unroll
        for (int t = 0; t < 4; ++t) {
            int n = ns * 64 + t * 16 + nl;     // B column (output feature)
            v16h bfr = load_b_frag(W + (size_t)n * HID + k + half * 16);
            acc[0][t] = wmma_f16(a0, bfr, acc[0][t]);
            acc[1][t] = wmma_f16(a1, bfr, acc[1][t]);
        }
    }

    const int mbase = half * 8;
#pragma unroll
    for (int t = 0; t < 4; ++t) {
        int n = ns * 64 + t * 16 + nl;
        float bv = bias[n];
        int hh = n >> 6;       // head
        int dk = n & 63;
#pragma unroll
        for (int g = 0; g < 2; ++g) {
#pragma unroll
            for (int r = 0; r < 8; ++r) {
                int row = mt * 32 + g * 16 + mbase + r;
                float v = acc[g][t][r] + bv;
                if (mode == 2) {
                    out_f[(size_t)row * HID + n] = v;
                } else {
                    int b = row >> 11;             // / SS
                    int s = row & (SS - 1);
                    if (mode == 0)
                        out_h[(((size_t)(b * NH + hh) * SS) + s) * DKD + dk] =
                            (_Float16)v;
                    else
                        out_h[(((size_t)(b * NH + hh) * DKD) + dk) * SS + s] =
                            (_Float16)v;
                }
            }
        }
    }
}

// ---------------------------------------------------------------- flash attention
// Q,K: [B*H, S, DK] f16 ; VT: [B*H, DK, S] f16 ; ctx out: [B, S, HID] f16
// Online softmax kept in base-2 so v_exp_f32 is used directly (exp2).
__global__ __launch_bounds__(256) void attn_wmma(
    const _Float16* __restrict__ Q, const _Float16* __restrict__ K,
    const _Float16* __restrict__ VT, _Float16* __restrict__ ctx) {
    __shared__ __attribute__((aligned(16))) _Float16 lds[8][16 * 32];

    const int lane = threadIdx.x & 31;
    const int wid  = threadIdx.x >> 5;
    const int qt   = blockIdx.x * 8 + wid;     // 128 q-tiles per (b,h)
    const int bh   = blockIdx.y;               // 64
    const int half = lane >> 4;
    const int nl   = lane & 15;
    _Float16* pt = lds[wid];

    const _Float16* qb = Q  + (size_t)bh * SS * DKD;
    const _Float16* kb = K  + (size_t)bh * SS * DKD;
    const _Float16* vb = VT + (size_t)bh * DKD * SS;

    // scores scaled into base-2 domain: s * (1/sqrt(64)) * log2(e)
    const float SCL = 0.125f * 1.44269504088896f;

    // Q A-fragments (row = query nl within tile), d = 0..63 -> 2 frags of K=32
    const _Float16* qrow = qb + (size_t)(qt * 16 + nl) * DKD;
    v16h qa[2];
#pragma unroll
    for (int f = 0; f < 2; ++f) qa[f] = load_a_frag(qrow, f * 32, half);

    v8f acc[4] = {};
    float mrun[8], lrun[8];
#pragma unroll
    for (int r = 0; r < 8; ++r) { mrun[r] = -1e30f; lrun[r] = 0.0f; }

    const int kend = qt * 16 + 16;             // causal: keys < kend
    const int nch  = (kend + 31) >> 5;         // 32-key chunks

    for (int c = 0; c < nch; ++c) {
        const int k0 = c * 32;

        // ---- scores: two 16x16 tiles, K-dim = d (64) => 2 WMMAs each
        v8f st[2] = {};
#pragma unroll
        for (int t = 0; t < 2; ++t) {
            int key = k0 + t * 16 + nl;        // B column = key index
            const _Float16* krow = kb + (size_t)key * DKD;
#pragma unroll
            for (int f = 0; f < 2; ++f)
                st[t] = wmma_f16(qa[f], load_b_frag(krow + f * 32 + half * 16),
                                 st[t]);
        }

        // ---- online softmax in D-register layout (row = half*8 + r)
        float mnew[8];
#pragma unroll
        for (int r = 0; r < 8; ++r) {
            int sq = qt * 16 + half * 8 + r;
            float x0 = st[0][r] * SCL;
            float x1 = st[1][r] * SCL;
            if (k0 + nl > sq)       x0 = -1e30f;
            if (k0 + 16 + nl > sq)  x1 = -1e30f;
            st[0][r] = x0; st[1][r] = x1;
            float mx = fmaxf(x0, x1);
            mx = fmaxf(mx, __shfl_xor(mx, 1, 32));
            mx = fmaxf(mx, __shfl_xor(mx, 2, 32));
            mx = fmaxf(mx, __shfl_xor(mx, 4, 32));
            mx = fmaxf(mx, __shfl_xor(mx, 8, 32));
            mnew[r] = fmaxf(mrun[r], mx);
        }
#pragma unroll
        for (int r = 0; r < 8; ++r) {
            float p0 = exp2f(st[0][r] - mnew[r]);
            float p1 = exp2f(st[1][r] - mnew[r]);
            st[0][r] = p0; st[1][r] = p1;
            float rs = p0 + p1;
            rs += __shfl_xor(rs, 1, 32);
            rs += __shfl_xor(rs, 2, 32);
            rs += __shfl_xor(rs, 4, 32);
            rs += __shfl_xor(rs, 8, 32);
            float sc = exp2f(mrun[r] - mnew[r]);
            lrun[r] = lrun[r] * sc + rs;
            mrun[r] = mnew[r];
#pragma unroll
            for (int t = 0; t < 4; ++t) acc[t][r] *= sc;
        }

        // ---- transpose P (D layout) -> A fragment via per-wave LDS tile
#pragma unroll
        for (int r = 0; r < 8; ++r) {
            int m = half * 8 + r;
            pt[m * 32 + nl]      = (_Float16)st[0][r];
            pt[m * 32 + 16 + nl] = (_Float16)st[1][r];
        }
        asm volatile("s_wait_dscnt 0" ::: "memory");   // wave-internal W->R
        v16h pa = load_a_frag(pt + nl * 32, 0, half);

        // ---- PV: out[16q x 64d] += P[16x32] * V[32x64]; VT rows contiguous
#pragma unroll
        for (int t = 0; t < 4; ++t) {
            const _Float16* vrow =
                vb + (size_t)(t * 16 + nl) * SS + k0 + half * 16;
            acc[t] = wmma_f16(pa, load_b_frag(vrow), acc[t]);
        }
    }

    // ---- normalize + store ctx as [b, s, h*64 + d] f16
    const int b = bh >> 4, hh = bh & 15;
#pragma unroll
    for (int t = 0; t < 4; ++t) {
        int e = hh * DKD + t * 16 + nl;
#pragma unroll
        for (int r = 0; r < 8; ++r) {
            int s = qt * 16 + half * 8 + r;
            float v = acc[t][r] / lrun[r];
            ctx[((size_t)(b * SS + s)) * HID + e] = (_Float16)v;
        }
    }
}

// ---------------------------------------------------------------- launch
extern "C" void kernel_launch(void* const* d_in, const int* in_sizes, int n_in,
                              void* d_out, int out_size, void* d_ws,
                              size_t ws_size, hipStream_t stream) {
    const float* x  = (const float*)d_in[0];
    // d_in[1] = attn_mask (causal tril) -> computed analytically
    const float* wq = (const float*)d_in[2];
    const float* bq = (const float*)d_in[3];
    const float* wk = (const float*)d_in[4];
    const float* bk = (const float*)d_in[5];
    const float* wv = (const float*)d_in[6];
    const float* bv = (const float*)d_in[7];
    const float* wo = (const float*)d_in[8];
    const float* bo = (const float*)d_in[9];
    float* out = (float*)d_out;

    char* ws = (char*)d_ws;
    size_t off = 0;
    auto carve = [&](size_t elems) {
        _Float16* p = (_Float16*)(ws + off);
        off += elems * sizeof(_Float16);
        return p;
    };
    const size_t NX = (size_t)ROWS * HID;   // 8388608
    const size_t NW = (size_t)HID * HID;    // 1048576
    _Float16* xb  = carve(NX);
    _Float16* wqh = carve(NW);
    _Float16* wkh = carve(NW);
    _Float16* wvh = carve(NW);
    _Float16* woh = carve(NW);
    _Float16* qbf = carve(NX);
    _Float16* kbf = carve(NX);
    _Float16* vtb = carve(NX);
    _Float16* ctx = carve(NX);

    // convert inputs to f16
    cvt_f32_to_f16<<<4096, 256, 0, stream>>>(x, xb, (int)NX);
    cvt_f32_to_f16<<<1024, 256, 0, stream>>>(wq, wqh, (int)NW);
    cvt_f32_to_f16<<<1024, 256, 0, stream>>>(wk, wkh, (int)NW);
    cvt_f32_to_f16<<<1024, 256, 0, stream>>>(wv, wvh, (int)NW);
    cvt_f32_to_f16<<<1024, 256, 0, stream>>>(wo, woh, (int)NW);

    // QKV projections: 256 m-tiles * 16 n-strips / 8 waves = 512 blocks
    gemm_wmma<<<512, 256, 0, stream>>>(xb, wqh, bq, qbf, nullptr, 0);
    gemm_wmma<<<512, 256, 0, stream>>>(xb, wkh, bk, kbf, nullptr, 0);
    gemm_wmma<<<512, 256, 0, stream>>>(xb, wvh, bv, vtb, nullptr, 1);

    // flash attention: 128 q-tiles / 8 waves = 16 blocks x, 64 (b,h) y
    attn_wmma<<<dim3(16, 64), 256, 0, stream>>>(qbf, kbf, vtb, ctx);

    // output projection -> fp32 result
    gemm_wmma<<<512, 256, 0, stream>>>(ctx, woh, bo, nullptr, out, 2);
}
